// MACEMetaEncoder_16819091931682
// MI455X (gfx1250) — matrix operand order repaired and checked by reading, not compiled, not run on hardware
//
#include <hip/hip_runtime.h>
#include <hip/hip_bf16.h>
#include <math.h>

#define Ncnt 50000
#define Ecnt 1600000
#define HSc 64
#define HVc 32
#define NBc 32
#define CUTF 5.0f
#define SQRT3F 1.7320508075688772f
#define C2F 2.7386127875258306f

typedef __attribute__((ext_vector_type(16))) _Float16 v16h;
typedef __attribute__((ext_vector_type(8)))  float    v8f;

union Frag { v16h h; unsigned u[8]; };

__device__ __forceinline__ unsigned pk(float a, float b) {
  union { unsigned u; _Float16 h[2]; } t;
  t.h[0] = (_Float16)a; t.h[1] = (_Float16)b;
  return t.u;
}

__device__ __forceinline__ float fixv(float x) {
  if (isnan(x)) return 0.0f;
  if (isinf(x)) return x > 0.f ? 3.4028235e38f : -3.4028235e38f;
  return x;
}

__device__ __forceinline__ v8f wmma16(v16h a, v16h b, v8f c) {
  // D = A(16x32 f16) x B(32x16 f16) + C(16x16 f32)
  return __builtin_amdgcn_wmma_f32_16x16x32_f16(
      false, a, false, b, (short)0, c, false, false);
}

// A fragment (16x32 f16) from LDS rows stored as packed f16 pairs (dwords).
// 16-bit A layout: lanes0-15 K={0..7,16..23}, lanes16-31 K={8..15,24..31}.
__device__ __forceinline__ Frag loadA_pairs(const unsigned* rows, int rowDw,
                                            int kt, int lane) {
  Frag a;
  int m   = lane & 15;
  int off = (lane & 16) ? 8 : 0;
#pragma unroll
  for (int v = 0; v < 8; ++v) {
    int k = kt * 32 + off + ((v & 4) ? 16 : 0) + 2 * (v & 3);
    a.u[v] = rows[m * rowDw + (k >> 1)];
  }
  return a;
}

// A fragment from f32 LDS rows with element stride (for [c*3+d] layouts).
__device__ __forceinline__ Frag loadA_strided(const float* rows, int rowLen,
                                              int stride, int d0, int lane) {
  Frag a;
  int m   = lane & 15;
  int off = (lane & 16) ? 8 : 0;
#pragma unroll
  for (int v = 0; v < 8; ++v) {
    int k  = off + ((v & 4) ? 16 : 0) + 2 * (v & 3);
    float f0 = rows[m * rowLen + k * stride + d0];
    float f1 = rows[m * rowLen + (k + 1) * stride + d0];
    a.u[v] = pk(f0, f1);
  }
  return a;
}

// B fragment from pre-packed weight buffer: frag = 32 lanes x 8 dwords.
__device__ __forceinline__ Frag loadB(const unsigned* __restrict__ wfrag,
                                      int fragIdx, int lane) {
  Frag b;
  const unsigned* p = wfrag + fragIdx * 256 + lane * 8;
#pragma unroll
  for (int v = 0; v < 8; ++v) b.u[v] = p[v];
  return b;
}

// Fragment index map (28 frags per layer):
//  [0,8): Ws (nt*2+kt)  [8,10): Wv (nt)  [10,14): Uvs (nt)
//  [14,18): Usv (nt*2+kt)  [18,26): Wos (nt*2+kt)  [26,28): Wov (nt)
__global__ void prep_weights(const float* __restrict__ Ws,
                             const float* __restrict__ Wv,
                             const float* __restrict__ Uvs,
                             const float* __restrict__ Usv,
                             const float* __restrict__ Wos,
                             const float* __restrict__ Wov,
                             unsigned* __restrict__ wfrag) {
  int wave = blockIdx.x;
  int lane = threadIdx.x;
  int l = wave / 28, fid = wave % 28;
  const float* W; int NN, nt, kt;
  if (fid < 8)        { W = Ws  + l * 64 * 64; NN = 64; nt = fid / 2;  kt = fid & 1; }
  else if (fid < 10)  { W = Wv  + l * 32 * 32; NN = 32; nt = fid - 8;  kt = 0; }
  else if (fid < 14)  { W = Uvs + l * 32 * 64; NN = 64; nt = fid - 10; kt = 0; }
  else if (fid < 18)  { W = Usv + l * 64 * 32; NN = 32; int x = fid - 14; nt = x / 2; kt = x & 1; }
  else if (fid < 26)  { W = Wos + l * 64 * 64; NN = 64; int x = fid - 18; nt = x / 2; kt = x & 1; }
  else                { W = Wov + l * 32 * 32; NN = 32; nt = fid - 26; kt = 0; }
  int n = nt * 16 + (lane & 15);
  int kbase = kt * 32 + ((lane & 16) ? 8 : 0);
  unsigned* out = wfrag + wave * 256 + lane * 8;
#pragma unroll
  for (int v = 0; v < 8; ++v) {
    int k = kbase + ((v & 4) ? 16 : 0) + 2 * (v & 3);
    out[v] = pk(W[k * NN + n], W[(k + 1) * NN + n]);
  }
}

__global__ void init_nodes(const float* __restrict__ s,
                           const float* __restrict__ v,
                           float* __restrict__ xs, float* __restrict__ xv) {
  int i = blockIdx.x * blockDim.x + threadIdx.x;
  if (i < Ncnt * 64) xs[i] = fixv(s[i]);
  if (i < Ncnt * 96) xv[i] = fixv(v[i]);
}

__global__ void zero_agg(float* __restrict__ as_, float* __restrict__ av_) {
  int i = blockIdx.x * blockDim.x + threadIdx.x;
  if (i < Ncnt * 64) as_[i] = 0.0f;
  if (i < Ncnt * 96) av_[i] = 0.0f;
}

#define NWPB 2  // waves (tiles) per block for node kernels

// sp = xs @ Ws[l];  vp[n,h,d] = sum_c xv[n,c,d]*Wv[l][c,h]
__global__ void node_in(const float* __restrict__ xs,
                        const float* __restrict__ xv,
                        const unsigned* __restrict__ wfrag, int layer,
                        float* __restrict__ sp, float* __restrict__ vp) {
  __shared__ unsigned s16[NWPB][16][32];
  __shared__ float    vf[NWPB][16][96];
  int lane = threadIdx.x & 31, w = threadIdx.x >> 5;
  int tile = blockIdx.x * NWPB + w;
  if (tile > 3124) tile = 3124;
  int n0 = tile * 16;
  int m = lane & 15, hi = lane >> 4;

  {  // stage scalar row (f32 -> packed f16) and vector row (f32)
    const float* row = xs + (size_t)(n0 + m) * 64 + hi * 32;
#pragma unroll
    for (int j = 0; j < 32; j += 4) {
      float4 f = *(const float4*)(row + j);
      s16[w][m][(hi * 32 + j) >> 1]       = pk(f.x, f.y);
      s16[w][m][((hi * 32 + j) >> 1) + 1] = pk(f.z, f.w);
    }
    const float* rv = xv + (size_t)(n0 + m) * 96 + hi * 48;
#pragma unroll
    for (int j = 0; j < 48; j += 4)
      *(float4*)&vf[w][m][hi * 48 + j] = *(const float4*)(rv + j);
  }
  __syncthreads();

  int L28 = layer * 28;
  Frag a0 = loadA_pairs(&s16[w][0][0], 32, 0, lane);
  Frag a1 = loadA_pairs(&s16[w][0][0], 32, 1, lane);
#pragma unroll
  for (int nt = 0; nt < 4; ++nt) {
    v8f c = {};
    c = wmma16(a0.h, loadB(wfrag, L28 + nt * 2 + 0, lane).h, c);
    c = wmma16(a1.h, loadB(wfrag, L28 + nt * 2 + 1, lane).h, c);
    int n = nt * 16 + m;
#pragma unroll
    for (int v = 0; v < 8; ++v) {
      int node = n0 + v + hi * 8;
      sp[(size_t)node * 64 + n] = c[v];
    }
  }
#pragma unroll
  for (int d0 = 0; d0 < 3; ++d0) {
    Frag a = loadA_strided(&vf[w][0][0], 96, 3, d0, lane);
#pragma unroll
    for (int nt = 0; nt < 2; ++nt) {
      v8f c = {};
      c = wmma16(a.h, loadB(wfrag, L28 + 8 + nt, lane).h, c);
      int h = nt * 16 + m;
#pragma unroll
      for (int v = 0; v < 8; ++v) {
        int node = n0 + v + hi * 8;
        vp[(size_t)node * 96 + h * 3 + d0] = c[v];
      }
    }
  }
}

// xs' = agg_s @ Wos[l];  xv'[n,g,d] = sum_h agg_v[n,h,d]*Wov[l][h,g]; nan_to_num
__global__ void node_out(const float* __restrict__ as_,
                         const float* __restrict__ av_,
                         const unsigned* __restrict__ wfrag, int layer,
                         float* __restrict__ xs, float* __restrict__ xv,
                         float* __restrict__ out, int final_) {
  __shared__ unsigned s16[NWPB][16][32];
  __shared__ float    vf[NWPB][16][96];
  int lane = threadIdx.x & 31, w = threadIdx.x >> 5;
  int tile = blockIdx.x * NWPB + w;
  if (tile > 3124) tile = 3124;
  int n0 = tile * 16;
  int m = lane & 15, hi = lane >> 4;

  {
    const float* row = as_ + (size_t)(n0 + m) * 64 + hi * 32;
#pragma unroll
    for (int j = 0; j < 32; j += 4) {
      float4 f = *(const float4*)(row + j);
      s16[w][m][(hi * 32 + j) >> 1]       = pk(f.x, f.y);
      s16[w][m][((hi * 32 + j) >> 1) + 1] = pk(f.z, f.w);
    }
    const float* rv = av_ + (size_t)(n0 + m) * 96 + hi * 48;
#pragma unroll
    for (int j = 0; j < 48; j += 4)
      *(float4*)&vf[w][m][hi * 48 + j] = *(const float4*)(rv + j);
  }
  __syncthreads();

  int L28 = layer * 28;
  Frag a0 = loadA_pairs(&s16[w][0][0], 32, 0, lane);
  Frag a1 = loadA_pairs(&s16[w][0][0], 32, 1, lane);
#pragma unroll
  for (int nt = 0; nt < 4; ++nt) {
    v8f c = {};
    c = wmma16(a0.h, loadB(wfrag, L28 + 18 + nt * 2 + 0, lane).h, c);
    c = wmma16(a1.h, loadB(wfrag, L28 + 18 + nt * 2 + 1, lane).h, c);
    int n = nt * 16 + m;
#pragma unroll
    for (int v = 0; v < 8; ++v) {
      int node = n0 + v + hi * 8;
      float val = fixv(c[v]);
      if (final_) out[(size_t)node * 160 + n] = val;
      else        xs[(size_t)node * 64 + n]  = val;
    }
  }
#pragma unroll
  for (int d0 = 0; d0 < 3; ++d0) {
    Frag a = loadA_strided(&vf[w][0][0], 96, 3, d0, lane);
#pragma unroll
    for (int nt = 0; nt < 2; ++nt) {
      v8f c = {};
      c = wmma16(a.h, loadB(wfrag, L28 + 26 + nt, lane).h, c);
      int h = nt * 16 + m;
#pragma unroll
      for (int v = 0; v < 8; ++v) {
        int node = n0 + v + hi * 8;
        float val = fixv(c[v]);
        if (final_) out[(size_t)node * 160 + 64 + h * 3 + d0] = val;
        else        xv[(size_t)node * 96 + h * 3 + d0]        = val;
      }
    }
  }
}

#define EWPB 2  // waves (16-edge tiles) per block for the edge kernel

__global__ void edge_pass(const int* __restrict__ ei,
                          const float* __restrict__ pos,
                          const float* __restrict__ shifts,
                          const float* __restrict__ Wr,
                          const float* __restrict__ Wrb,
                          const float* __restrict__ sp,
                          const float* __restrict__ vp,
                          const unsigned* __restrict__ wfrag, int layer,
                          float* __restrict__ as_, float* __restrict__ av_) {
  __shared__ float    wrs[NBc * 5 + 5];
  __shared__ unsigned seh[EWPB][16][32];   // se rows, packed f16
  __shared__ float    vef[EWPB][16][96];   // ve rows, f32
  __shared__ unsigned dth[EWPB][16][16];   // dot rows, packed f16
  __shared__ float    svf[EWPB][16][32];   // sv rows, f32
  __shared__ float    dls[EWPB][16][3];    // unit edge dir
  __shared__ float    wls[EWPB][16][5];    // gate-folded radial weights
  __shared__ int      dsti[EWPB][16];
  __shared__ int      srci[EWPB][16];

  for (int i = threadIdx.x; i < NBc * 5 + 5; i += blockDim.x)
    wrs[i] = (i < NBc * 5) ? Wr[layer * NBc * 5 + i]
                           : Wrb[layer * 5 + (i - NBc * 5)];
  __syncthreads();

  int lane = threadIdx.x & 31, w = threadIdx.x >> 5;
  int tile = blockIdx.x * EWPB + w;  // E/16 = 100000 tiles exactly
  int e0 = tile * 16;
  int m = lane & 15, hi = lane >> 4;

  // ---- phase 0: per-edge geometry + radial weights (gate folded in) ----
  if (lane < 16) {
    int e = e0 + lane;
    int src = ei[e], dst = ei[Ecnt + e];
    srci[w][lane] = src; dsti[w][lane] = dst;

    // warm L2->L0 path for the rows phase 1 will gather (global_prefetch_b8)
    {
      const float* ps = sp + (size_t)src * 64;
      const float* pv = vp + (size_t)src * 96;
      __builtin_prefetch(ps, 0, 3);
      __builtin_prefetch(ps + 32, 0, 3);
      __builtin_prefetch(pv, 0, 3);
      __builtin_prefetch(pv + 32, 0, 3);
      __builtin_prefetch(pv + 64, 0, 3);
    }

    float ex = pos[dst * 3 + 0] - pos[src * 3 + 0] + shifts[(size_t)e * 3 + 0];
    float ey = pos[dst * 3 + 1] - pos[src * 3 + 1] + shifts[(size_t)e * 3 + 1];
    float ez = pos[dst * 3 + 2] - pos[src * 3 + 2] + shifts[(size_t)e * 3 + 2];
    float r  = sqrtf(ex * ex + ey * ey + ez * ez);
    float inv = 1.0f / fmaxf(r, 1e-8f);
    dls[w][lane][0] = ex * inv;
    dls[w][lane][1] = ey * inv;
    dls[w][lane][2] = ez * inv;
    float u = r * (1.0f / CUTF);
    float gate = 0.0f;
    if (u < 1.0f) { float inner = 1.0f - u * u; gate = __expf(1.0f - 1.0f / inner); }
    float acc0 = 0.f, acc1 = 0.f, acc2 = 0.f, acc3 = 0.f, acc4 = 0.f;
#pragma unroll
    for (int b = 0; b < NBc; ++b) {
      float t  = (r - (CUTF * b / 31.0f)) * (32.0f / CUTF);
      float rb = __expf(-0.5f * t * t);
      acc0 += rb * wrs[b * 5 + 0];
      acc1 += rb * wrs[b * 5 + 1];
      acc2 += rb * wrs[b * 5 + 2];
      acc3 += rb * wrs[b * 5 + 3];
      acc4 += rb * wrs[b * 5 + 4];
    }
    wls[w][lane][0] = gate * (gate * acc0 + wrs[NBc * 5 + 0]);
    wls[w][lane][1] = gate * (gate * acc1 + wrs[NBc * 5 + 1]);
    wls[w][lane][2] = gate * (gate * acc2 + wrs[NBc * 5 + 2]);
    wls[w][lane][3] = gate * (gate * acc3 + wrs[NBc * 5 + 3]);
    wls[w][lane][4] = gate * (gate * acc4 + wrs[NBc * 5 + 4]);
  }
  __syncthreads();

  // ---- phase 1: gather se (f16 convert) / ve (async copy direct to LDS) ----
  {
    int src = srci[w][m];

    // ve rows: f32 copy with no conversion -> async global->LDS path
    // (GLOBAL_LOAD_ASYNC_TO_LDS_B128, tracked on ASYNCcnt)
    {
      const float* rv = vp + (size_t)src * 96 + hi * 48;
#pragma unroll
      for (int j = 0; j < 48; j += 4) {
        unsigned lds_addr = (unsigned)(size_t)&vef[w][m][hi * 48 + j];
        unsigned long long ga = (unsigned long long)(const void*)(rv + j);
        asm volatile("global_load_async_to_lds_b128 %0, %1, off"
                     :: "v"(lds_addr), "v"(ga) : "memory");
      }
    }

    // se row: needs f32->f16 packing, goes through VGPRs
    const float* row = sp + (size_t)src * 64 + hi * 32;
#pragma unroll
    for (int j = 0; j < 32; j += 4) {
      float4 f = *(const float4*)(row + j);
      seh[w][m][(hi * 32 + j) >> 1]       = pk(f.x, f.y);
      seh[w][m][((hi * 32 + j) >> 1) + 1] = pk(f.z, f.w);
    }

    asm volatile("s_wait_asynccnt 0x0" ::: "memory");
  }
  __syncthreads();

  // ---- phase 2: dot[e,h] = sqrt3 * (ve[h] . d) ----
  {
    float dx = dls[w][m][0], dy = dls[w][m][1], dz = dls[w][m][2];
    _Float16* drow = (_Float16*)&dth[w][m][0];
#pragma unroll
    for (int t = 0; t < 16; ++t) {
      int h = hi * 16 + t;
      float dv = SQRT3F * (vef[w][m][h * 3 + 0] * dx +
                           vef[w][m][h * 3 + 1] * dy +
                           vef[w][m][h * 3 + 2] * dz);
      drow[h] = (_Float16)dv;
    }
  }
  __syncthreads();

  int L28 = layer * 28;

  // ---- phase 3: m_s = w0'*se + w1'*(dot @ Uvs), scatter-add to agg_s ----
  Frag ad = loadA_pairs(&dth[w][0][0], 16, 0, lane);
#pragma unroll
  for (int nt = 0; nt < 4; ++nt) {
    v8f c = {};
    c = wmma16(ad.h, loadB(wfrag, L28 + 10 + nt, lane).h, c);
    int n = nt * 16 + m;
#pragma unroll
    for (int v = 0; v < 8; ++v) {
      int mm = v + hi * 8;
      union { unsigned u; _Float16 h2[2]; } t;
      t.u = seh[w][mm][n >> 1];
      float sev = (float)t.h2[n & 1];
      float val = wls[w][mm][1] * c[v] + wls[w][mm][0] * sev;
      atomicAdd(&as_[(size_t)dsti[w][mm] * 64 + n], val);
    }
  }

  // ---- phase 4: sv = se @ Usv -> LDS ----
  Frag a0 = loadA_pairs(&seh[w][0][0], 32, 0, lane);
  Frag a1 = loadA_pairs(&seh[w][0][0], 32, 1, lane);
#pragma unroll
  for (int nt = 0; nt < 2; ++nt) {
    v8f c = {};
    c = wmma16(a0.h, loadB(wfrag, L28 + 14 + nt * 2 + 0, lane).h, c);
    c = wmma16(a1.h, loadB(wfrag, L28 + 14 + nt * 2 + 1, lane).h, c);
#pragma unroll
    for (int v = 0; v < 8; ++v)
      svf[w][v + hi * 8][nt * 16 + m] = c[v];
  }
  __syncthreads();

  // ---- phase 5: m_v = alpha[h]*d + beta*ve, scatter-add to agg_v ----
  // alpha = sqrt3*w2'*sv + (C2/sqrt3)*w4'*dot ; beta = w3' - (C2/3)*w4'
  {
    float dx = dls[w][m][0], dy = dls[w][m][1], dz = dls[w][m][2];
    float w2 = wls[w][m][2], w3 = wls[w][m][3], w4 = wls[w][m][4];
    float beta = w3 - (C2F / 3.0f) * w4;
    float ca   = C2F / SQRT3F;
    float* avb = av_ + (size_t)dsti[w][m] * 96;
    const _Float16* drow = (const _Float16*)&dth[w][m][0];
#pragma unroll
    for (int t = 0; t < 16; ++t) {
      int h = hi * 16 + t;
      float alpha = SQRT3F * w2 * svf[w][m][h] + ca * w4 * (float)drow[h];
      atomicAdd(&avb[h * 3 + 0], alpha * dx + beta * vef[w][m][h * 3 + 0]);
      atomicAdd(&avb[h * 3 + 1], alpha * dy + beta * vef[w][m][h * 3 + 1]);
      atomicAdd(&avb[h * 3 + 2], alpha * dz + beta * vef[w][m][h * 3 + 2]);
    }
  }
}

// ---- workspace layout (bytes, all 256-aligned) ----
#define OFF_WF 0u
#define OFF_XS 86016u
#define OFF_XV (OFF_XS + 12800000u)
#define OFF_SP (OFF_XV + 19200000u)
#define OFF_VP (OFF_SP + 12800000u)
#define OFF_AS (OFF_VP + 19200000u)
#define OFF_AV (OFF_AS + 12800000u)

extern "C" void kernel_launch(void* const* d_in, const int* in_sizes, int n_in,
                              void* d_out, int out_size, void* d_ws, size_t ws_size,
                              hipStream_t stream) {
  const float* pos    = (const float*)d_in[0];
  const float* sfeat  = (const float*)d_in[1];
  const float* vfeat  = (const float*)d_in[2];
  const float* shifts = (const float*)d_in[3];
  const int*   ei     = (const int*)d_in[4];
  const float* Ws     = (const float*)d_in[5];
  const float* Wv     = (const float*)d_in[6];
  const float* Uvs    = (const float*)d_in[7];
  const float* Usv    = (const float*)d_in[8];
  const float* Wr     = (const float*)d_in[9];
  const float* Wrb    = (const float*)d_in[10];
  const float* Wos    = (const float*)d_in[11];
  const float* Wov    = (const float*)d_in[12];
  float* out = (float*)d_out;
  char*  ws  = (char*)d_ws;

  unsigned* wfrag = (unsigned*)(ws + OFF_WF);
  float* xs = (float*)(ws + OFF_XS);
  float* xv = (float*)(ws + OFF_XV);
  float* sp = (float*)(ws + OFF_SP);
  float* vp = (float*)(ws + OFF_VP);
  float* as_ = (float*)(ws + OFF_AS);
  float* av_ = (float*)(ws + OFF_AV);

  prep_weights<<<84, 32, 0, stream>>>(Ws, Wv, Uvs, Usv, Wos, Wov, wfrag);
  init_nodes<<<(Ncnt * 96 + 255) / 256, 256, 0, stream>>>(sfeat, vfeat, xs, xv);

  const int nodeBlocks = (3125 + NWPB - 1) / NWPB;   // 1563
  const int edgeBlocks = (Ecnt / 16) / EWPB;         // 50000

  for (int l = 0; l < 3; ++l) {
    node_in<<<nodeBlocks, 32 * NWPB, 0, stream>>>(xs, xv, wfrag, l, sp, vp);
    zero_agg<<<(Ncnt * 96 + 255) / 256, 256, 0, stream>>>(as_, av_);
    edge_pass<<<edgeBlocks, 32 * EWPB, 0, stream>>>(ei, pos, shifts, Wr, Wrb,
                                                    sp, vp, wfrag, l, as_, av_);
    node_out<<<nodeBlocks, 32 * NWPB, 0, stream>>>(as_, av_, wfrag, l, xs, xv,
                                                   out, (l == 2) ? 1 : 0);
  }
}
